// KPConv_2173253452285
// MI455X (gfx1250) — matrix operand order repaired and checked by reading, not compiled, not run on hardware
//
#include <hip/hip_runtime.h>

// ---------------------------------------------------------------------------
// KPConv for MI455X (gfx1250, wave32), v3.
//
// Stage 1 (per point): weighted(16x64) = wT(16x32,f16) x featsT(32x64,f16)
//                      via 4x v_wmma_f32_16x16x32_f16 (operands in LDS)
// Stage 2 (per 16-pt tile): OUT(16x64) = Weighted(16x1024) x W2(1024x64)
//                      via 32 chunks x 4 tiles of v_wmma_f32_16x16x32_f16,
//                      B operands streamed from a pre-packed fragment-ordered
//                      f16 weight table in global memory (128KB, L2-resident,
//                      one aligned 32B coalesced load per lane per fragment).
//
// v3: LDS row strides padded so all WMMA operand ds_load_b128s are
//     bank-conflict-free (weighted: 516 dw ≡ 4 mod 64; wT/featsT: 20 dw).
// LDS = 158KB/block -> 2 blocks (8 waves) resident per WGP.
// ---------------------------------------------------------------------------

typedef __attribute__((ext_vector_type(16))) _Float16 v16h;
typedef __attribute__((ext_vector_type(8)))  float    v8f;

#define M_PTS 50000
#define N_PTS 100000
#define NH    32      // neighbors per point (== wave32 lanes)
#define NK    15      // kernel points
#define NKP   16      // padded K (row 15 zeroed)
#define CIN   64
#define COUT  64
#define KC    1024    // contraction axis: 16 k-rows * 64 c
#define KCP   1032    // padded per-point stride (halves): 2064B=516dw ≡ 4 mod 64
#define FSTR  40      // featsT row stride (halves): 80B = 20 dwords
#define WSTR  40      // wT     row stride (halves): 80B = 20 dwords
#define NCHUNK (KC / 32)                       // 32 stage-2 k-chunks
#define W2FRAG_HALVES (NCHUNK * 4 * 32 * 16)   // 65536 halves = 128KB

#define WAVES 4
#define TPB   (WAVES * 32)
#define PTS_PER_WAVE  16
#define PTS_PER_BLOCK (WAVES * PTS_PER_WAVE)

// ---- dynamic LDS layout (bytes) ----
#define LDS_KP_BYTES       256                        // kernel points (45 f32)
#define LDS_WAVE_OFF       LDS_KP_BYTES
#define LDS_WEIGHTED_BYTES (PTS_PER_WAVE * KCP * 2)   // 33024: weighted[p][kc]
#define LDS_FEATS_BYTES    (CIN * FSTR * 2)           // 5120:  featsT[c][h]
#define LDS_WT_BYTES       (NKP * WSTR * 2)           // 1280:  wT[k][h]
#define LDS_CNT_BYTES      64                         // per-point nbr counts
#define LDS_PER_WAVE  (LDS_WEIGHTED_BYTES + LDS_FEATS_BYTES + LDS_WT_BYTES + LDS_CNT_BYTES)
#define LDS_TOTAL     (LDS_WAVE_OFF + WAVES * LDS_PER_WAVE)   // 158208 < 160KB

// ---------------------------------------------------------------------------
// One-time pre-pack: weights (15,64,64 f32) -> fragment-ordered f16 table.
// frag[((chunk*4 + t)*32 + lane)*16 + e] holds W2T[d = t*16 + (lane&15)]
// [kc = chunk*32 + (lane<16 ? 0 : 16) + e], kc = k*64 + c, k==15 rows zero.
// ---------------------------------------------------------------------------
extern "C" __global__ __launch_bounds__(256, 4)
void kpconv_pack_w2(const float* __restrict__ weights, _Float16* __restrict__ frag)
{
    int j = blockIdx.x * 256 + threadIdx.x;        // 0 .. 65535
    int e     = j & 15;
    int lane  = (j >> 4) & 31;
    int ft    = j >> 9;                            // 0 .. 127
    int t     = ft & 3;
    int chunk = ft >> 2;
    int d  = t * 16 + (lane & 15);
    int kc = chunk * 32 + ((lane < 16) ? 0 : 16) + e;
    int k  = kc >> 6;
    int c  = kc & 63;
    float v = (k < NK) ? weights[(k * CIN + c) * COUT + d] : 0.0f;
    frag[j] = (_Float16)v;
}

extern "C" __global__ __launch_bounds__(TPB, 2)
void kpconv_wmma_gfx1250(const float* __restrict__ q_points,        // (M,3)
                         const float* __restrict__ s_points,        // (N,3)
                         const float* __restrict__ s_feats,         // (N,64)
                         const long long* __restrict__ nbr_idx,     // (M,32) i64
                         const _Float16* __restrict__ w2frag,       // packed (ws)
                         const float* __restrict__ kpts,            // (15,3)
                         float* __restrict__ out)                   // (M,64)
{
    extern __shared__ char smem[];
    float* kp = (float*)smem;

    const int tid  = threadIdx.x;
    const int wave = tid >> 5;
    const int lane = tid & 31;

    char* wbase = smem + LDS_WAVE_OFF + wave * LDS_PER_WAVE;
    _Float16* weighted = (_Float16*)(wbase);                                 // [p][kc]
    _Float16* featsT   = (_Float16*)(wbase + LDS_WEIGHTED_BYTES);            // [c][h]
    _Float16* wT       = (_Float16*)(wbase + LDS_WEIGHTED_BYTES + LDS_FEATS_BYTES); // [k][h]
    float*    cnts     = (float*)   (wbase + LDS_WEIGHTED_BYTES + LDS_FEATS_BYTES + LDS_WT_BYTES);

    if (tid < NK * 3) kp[tid] = kpts[tid];
    __syncthreads();

    const int tile0 = blockIdx.x * PTS_PER_BLOCK + wave * PTS_PER_WAVE;

    // WMMA fragment addressing (this code's contiguous k order, same for A and B):
    //  A 16x32 f16 : lanes 0-15 row M=lane  hold K 0-15
    //                lanes 16-31 row M=lane-16 hold K 16-31
    //  B 32x16 f16 : lanes 0-15 col N=lane  hold K 0-15
    //                lanes 16-31 col N=lane-16 hold K 16-31
    const int arow = lane & 15;
    const int akb  = (lane < 16) ? 0 : 16;
    const int bcol = lane & 15;
    const int bkb  = (lane < 16) ? 0 : 16;
    const int mrow = (lane < 16) ? 0 : 8;   // D fragment row base

    // Preload all 16 neighbor-index rows (lane == neighbor slot h) so we can
    // prefetch the next point's gather while doing the current point's WMMAs.
    long long idxs[PTS_PER_WAVE];
    #pragma unroll
    for (int p = 0; p < PTS_PER_WAVE; ++p) {
        int gp  = tile0 + p;
        int gpc = gp < M_PTS ? gp : (M_PTS - 1);
        idxs[p] = nbr_idx[(long long)gpc * NH + lane];
    }

    // ==================================================================
    // Stage 1: per point, build wT + featsT in LDS, 4 WMMAs -> weighted
    // ==================================================================
    for (int p = 0; p < PTS_PER_WAVE; ++p) {
        int gp  = tile0 + p;
        int gpc = gp < M_PTS ? gp : (M_PTS - 1);

        float qx = q_points[gpc * 3 + 0];
        float qy = q_points[gpc * 3 + 1];
        float qz = q_points[gpc * 3 + 2];

        long long idx   = idxs[p];
        bool      valid = (idx >= 0) && (idx < (long long)N_PTS);
        long long cidx  = valid ? idx : 0;

        // prefetch next point's feature row into cache (global_prefetch_b8)
        if (p + 1 < PTS_PER_WAVE) {
            long long nidx = idxs[p + 1];
            if (nidx >= 0 && nidx < (long long)N_PTS)
                __builtin_prefetch(s_feats + (size_t)nidx * CIN, 0, 3);
        }

        float nx = valid ? s_points[cidx * 3 + 0] : 1.0e6f;
        float ny = valid ? s_points[cidx * 3 + 1] : 1.0e6f;
        float nz = valid ? s_points[cidx * 3 + 2] : 1.0e6f;
        float rx = nx - qx, ry = ny - qy, rz = nz - qz;

        // kernel correlations -> wT[k][h] (row 15 zeroed: pads A to 16 rows)
        #pragma unroll
        for (int k = 0; k < NKP; ++k) {
            float w = 0.0f;
            if (k < NK) {
                float dx = rx - kp[k * 3 + 0];
                float dy = ry - kp[k * 3 + 1];
                float dz = rz - kp[k * 3 + 2];
                float d2 = dx * dx + dy * dy + dz * dz;
                w = fmaxf(1.0f - sqrtf(d2) * 0.5f, 0.0f);   // 1/SIGMA = 0.5
            }
            wT[k * WSTR + lane] = (_Float16)w;
        }

        // coalesced 256B feature-row gather (one row per lane), f32->f16,
        // stored transposed so B-operand reads are contiguous along h.
        const float4* frow = (const float4*)(s_feats + (size_t)cidx * CIN);
        float rowsum = 0.0f;
        #pragma unroll
        for (int j = 0; j < CIN / 4; ++j) {
            float4 f = frow[j];
            if (!valid) { f.x = 0.f; f.y = 0.f; f.z = 0.f; f.w = 0.f; }
            rowsum += f.x + f.y + f.z + f.w;
            featsT[(j * 4 + 0) * FSTR + lane] = (_Float16)f.x;
            featsT[(j * 4 + 1) * FSTR + lane] = (_Float16)f.y;
            featsT[(j * 4 + 2) * FSTR + lane] = (_Float16)f.z;
            featsT[(j * 4 + 3) * FSTR + lane] = (_Float16)f.w;
        }

        // neighbor count: reference rule sum(feats) > 0 (padding row sums to 0)
        unsigned long long bal = __ballot(rowsum > 0.0f);
        int nn = (int)__popcll(bal);
        if (nn < 1) nn = 1;
        if (lane == 0) cnts[p] = (float)nn;

        // A fragment: wT rows (k) x contraction h  (conflict-free b128 reads)
        v16h a;
        #pragma unroll
        for (int i = 0; i < 8; ++i) {
            a[i]     = wT[arow * WSTR + akb + i];
            a[8 + i] = wT[arow * WSTR + akb + 8 + i];
        }

        // 4 column tiles of featsT -> weighted[p][k*64+c]
        #pragma unroll
        for (int t = 0; t < 4; ++t) {
            int c0 = t * 16;
            v16h b;
            #pragma unroll
            for (int i = 0; i < 16; ++i)
                b[i] = featsT[(c0 + bcol) * FSTR + bkb + i];
            v8f acc = {};
            acc = __builtin_amdgcn_wmma_f32_16x16x32_f16(
                false, a, false, b, (short)0, acc, false, false);
            #pragma unroll
            for (int r = 0; r < 8; ++r)
                weighted[p * KCP + (mrow + r) * CIN + c0 + bcol] = (_Float16)acc[r];
        }
    }

    // ==================================================================
    // Stage 2: OUT(16x64) = Weighted(16x1024) x W2(1024x64)
    //          A from LDS (conflict-free), B streamed from the L2 table
    // ==================================================================
    const v16h* __restrict__ w2 = (const v16h*)w2frag;
    v8f o[4] = {v8f{}, v8f{}, v8f{}, v8f{}};
    for (int chunk = 0; chunk < NCHUNK; ++chunk) {
        int kc0 = chunk * 32;
        v16h a;
        #pragma unroll
        for (int i = 0; i < 8; ++i) {
            a[i]     = weighted[arow * KCP + kc0 + akb + i];
            a[8 + i] = weighted[arow * KCP + kc0 + akb + 8 + i];
        }
        #pragma unroll
        for (int t = 0; t < 4; ++t) {
            v16h b = w2[(chunk * 4 + t) * 32 + lane];   // one 32B coalesced load
            o[t] = __builtin_amdgcn_wmma_f32_16x16x32_f16(
                false, a, false, b, (short)0, o[t], false, false);
        }
    }

    // normalize + store (D fragment: lane holds col d = t*16 + bcol,
    // rows m = mrow + r over the 8 accumulator registers)
    #pragma unroll
    for (int t = 0; t < 4; ++t) {
        #pragma unroll
        for (int r = 0; r < 8; ++r) {
            int m  = mrow + r;
            int gp = tile0 + m;
            if (gp < M_PTS) {
                float inv = 1.0f / cnts[m];
                out[(size_t)gp * COUT + t * 16 + bcol] = o[t][r] * inv;
            }
        }
    }
}

extern "C" void kernel_launch(void* const* d_in, const int* in_sizes, int n_in,
                              void* d_out, int out_size, void* d_ws, size_t ws_size,
                              hipStream_t stream) {
    (void)in_sizes; (void)n_in; (void)ws_size; (void)out_size;

    const float*     q_points = (const float*)d_in[0];
    const float*     s_points = (const float*)d_in[1];
    const float*     s_feats  = (const float*)d_in[2];
    const long long* nbr_idx  = (const long long*)d_in[3];
    const float*     weights  = (const float*)d_in[4];
    const float*     kpts     = (const float*)d_in[5];
    float*           out      = (float*)d_out;
    _Float16*        frag     = (_Float16*)d_ws;       // 128KB packed weights

    // One-time (per launch) weight pre-pack into fragment order.
    kpconv_pack_w2<<<W2FRAG_HALVES / 256, 256, 0, stream>>>(weights, frag);

    // CDNA5 WGP allows 320KB LDS per workgroup; request our 158KB
    // (2 blocks -> 8 waves resident per WGP).
    (void)hipFuncSetAttribute((const void*)kpconv_wmma_gfx1250,
                              hipFuncAttributeMaxDynamicSharedMemorySize,
                              LDS_TOTAL);

    const int grid = (M_PTS + PTS_PER_BLOCK - 1) / PTS_PER_BLOCK;  // 782
    kpconv_wmma_gfx1250<<<grid, TPB, LDS_TOTAL, stream>>>(
        q_points, s_points, s_feats, nbr_idx, frag, kpts, out);
}
